// SAGE_29678224016205
// MI455X (gfx1250) — matrix operand order repaired and checked by reading, not compiled, not run on hardware
//
#include <hip/hip_runtime.h>
#include <stdint.h>

// ---------------------------------------------------------------------------
// GraphSAGE (3 layers) + 2 edge-MLPs for MI455X (gfx1250, wave32, WMMA).
// Node features kept in f16 (halves gather traffic on the 800K-edge graph);
// all GEMMs via v_wmma_f32_16x16x32_f16 with f32 accumulation; segment-sum
// via f32 global atomics; weights pre-packed into the WMMA B-fragment layout.
// ---------------------------------------------------------------------------

namespace {
constexpr int kNodes = 50000;
constexpr int kEdges = 800000;
constexpr int kF     = 128;   // IN_F == HID == OUT_F
constexpr int kEHid  = 32;
}

typedef _Float16 v16h __attribute__((ext_vector_type(16)));
typedef _Float16 h8   __attribute__((ext_vector_type(8)));
typedef _Float16 h4   __attribute__((ext_vector_type(4)));
typedef float    v8f  __attribute__((ext_vector_type(8)));

__device__ __forceinline__ v8f wmma_f16(v16h a, v16h b, v8f c) {
  // D = A(16x32 f16) * B(32x16 f16) + C(16x16 f32)
  return __builtin_amdgcn_wmma_f32_16x16x32_f16(false, a, false, b, (short)0, c,
                                                false, false);
}

// A-fragment (16x32 f16) from a row-major f16 row.
// Layout: lane holds row M = lane&15; elems 0..7 = K[kt*32 + hs*8 .. +7],
// elems 8..15 = K[kt*32 + 16 + hs*8 .. +7], hs = lane>>4.
__device__ __forceinline__ v16h load_a16(const _Float16* row, int kt, int hs) {
  const h8* p0 = reinterpret_cast<const h8*>(row + kt * 32 + hs * 8);
  const h8* p1 = reinterpret_cast<const h8*>(row + kt * 32 + 16 + hs * 8);
  h8 x = *p0, y = *p1;
  v16h r;
#pragma unroll
  for (int i = 0; i < 8; ++i) { r[i] = x[i]; r[i + 8] = y[i]; }
  return r;
}

// A-fragment from an f32 row, scaled by s (mean aggregation: agg * rdeg).
__device__ __forceinline__ v16h load_a32_scaled(const float* row, int kt, int hs,
                                                float s) {
  const float* p0 = row + kt * 32 + hs * 8;
  const float* p1 = row + kt * 32 + 16 + hs * 8;
  v16h r;
#pragma unroll
  for (int i = 0; i < 8; ++i) {
    r[i]     = (_Float16)(p0[i] * s);
    r[i + 8] = (_Float16)(p1[i] * s);
  }
  return r;
}

// ---------------------------------------------------------------------------
// Pack W[K x N] (f32, row-major) into WMMA B-fragment layout (f16, zero pad).
// B layout: chunk (ntile, kt) of 512 f16; lane = (n&15) | ((k>>3 & 1)<<4);
// element = (k&7) | ((k>>4 & 1)<<3). Lane fragment = 32 contiguous bytes -> one coalesced load.
// ---------------------------------------------------------------------------
__global__ void pack_b_f16(const float* __restrict__ W, _Float16* __restrict__ dst,
                           int K, int N, int Kpad, int Npad) {
  int idx = blockIdx.x * blockDim.x + threadIdx.x;
  int total = Kpad * Npad;
  if (idx >= total) return;
  int k = idx / Npad;
  int n = idx - k * Npad;
  float v = (k < K && n < N) ? W[k * N + n] : 0.0f;
  int ntile = n >> 4;
  int kt = k >> 5;
  int lane = (n & 15) | (((k >> 3) & 1) << 4);
  int elem = (k & 7) | (((k >> 4) & 1) << 3);
  int numKt = Kpad >> 5;
  dst[(size_t)(ntile * numKt + kt) * 512 + lane * 16 + elem] = (_Float16)v;
}

__global__ void cvt_f32_to_f16(const float* __restrict__ src,
                               _Float16* __restrict__ dst, int n4) {
  int i = blockIdx.x * blockDim.x + threadIdx.x;
  if (i >= n4) return;
  float4 v = reinterpret_cast<const float4*>(src)[i];
  h4 o;
  o[0] = (_Float16)v.x; o[1] = (_Float16)v.y;
  o[2] = (_Float16)v.z; o[3] = (_Float16)v.w;
  reinterpret_cast<h4*>(dst)[i] = o;
}

__global__ void degree_kernel(const int* __restrict__ dst, float* __restrict__ deg,
                              int E) {
  int i = blockIdx.x * blockDim.x + threadIdx.x;
  if (i < E) atomicAdd(&deg[dst[i]], 1.0f);
}

__global__ void rdeg_kernel(float* deg, int n) {
  int i = blockIdx.x * blockDim.x + threadIdx.x;
  if (i < n) deg[i] = 1.0f / fmaxf(deg[i], 1.0f);
}

// Copy edge_feat [E,4] into columns 0..3 of the [E,12] output edge features.
__global__ void seed_edge_out(const float* __restrict__ ef,
                              float* __restrict__ out_ef, int n) {
  int i = blockIdx.x * blockDim.x + threadIdx.x;
  if (i >= n) return;
  int e = i >> 2, c = i & 3;
  out_ef[(size_t)e * 12 + c] = ef[i];
}

// Segment-sum: one 128-thread block per edge, one f32 atomic per feature.
__global__ void scatter_mean_kernel(const _Float16* __restrict__ h16,
                                    const int* __restrict__ src,
                                    const int* __restrict__ dst,
                                    float* __restrict__ agg) {
  int e = blockIdx.x;
  int t = threadIdx.x;  // 0..127
  int s = src[e], d = dst[e];
  float v = (float)h16[(size_t)s * kF + t];
  atomicAdd(&agg[(size_t)d * kF + t], v);
}

// ---------------------------------------------------------------------------
// SAGE layer: out = [relu]( h @ Wself + (agg * rdeg) @ Wneigh + b )
// One wave per 16x16 output tile; 8 WMMAs over K = 128.
// C layout: elem j of v8f <-> (row M = j + 8*(lane>>4), col N = lane&15).
// ---------------------------------------------------------------------------
template <bool RELU, bool F16OUT>
__global__ void sage_wmma_kernel(const _Float16* __restrict__ hin,
                                 const float* __restrict__ agg,
                                 const float* __restrict__ rdeg,
                                 const _Float16* __restrict__ wself,
                                 const _Float16* __restrict__ wneigh,
                                 const float* __restrict__ bias,
                                 _Float16* __restrict__ out16,
                                 float* __restrict__ out32) {
  int wid  = threadIdx.x >> 5;
  int lane = threadIdx.x & 31;
  int gw = blockIdx.x * 4 + wid;      // 25000 waves total
  int mtile = gw >> 3;
  int ntile = gw & 7;
  int M = lane & 15, hs = lane >> 4, N = lane & 15;
  int row = mtile * 16 + M;
  float rd = rdeg[row];
  float bv = bias[ntile * 16 + N];
  v8f acc;
#pragma unroll
  for (int j = 0; j < 8; ++j) acc[j] = bv;

  const _Float16* hrow = hin + (size_t)row * kF;
  const float* arow = agg + (size_t)row * kF;
  const _Float16* bs = wself + (size_t)(ntile * 4) * 512 + lane * 16;
  const _Float16* bn = wneigh + (size_t)(ntile * 4) * 512 + lane * 16;
#pragma unroll
  for (int kt = 0; kt < 4; ++kt) {
    v16h a_s = load_a16(hrow, kt, hs);
    v16h a_n = load_a32_scaled(arow, kt, hs, rd);
    v16h b_s = *reinterpret_cast<const v16h*>(bs + kt * 512);
    v16h b_n = *reinterpret_cast<const v16h*>(bn + kt * 512);
    acc = wmma_f16(a_s, b_s, acc);
    acc = wmma_f16(a_n, b_n, acc);
  }
#pragma unroll
  for (int j = 0; j < 8; ++j) {
    float v = RELU ? fmaxf(acc[j], 0.0f) : acc[j];
    int r = mtile * 16 + j + 8 * hs;
    int c = ntile * 16 + N;
    if (F16OUT) out16[(size_t)r * kF + c] = (_Float16)v;
    else        out32[(size_t)r * kF + c] = v;
  }
}

// ---------------------------------------------------------------------------
// Edge MLP: e = relu(h[src]@Wu + h[dst]@Wv + ef@We + b1) @ W2 + b2
// One wave per 16 edges; hidden (16x32) bounced through LDS to re-enter as an
// A-fragment for the second GEMM (W2 padded to 32x16, only 4 cols stored).
// ---------------------------------------------------------------------------
__global__ void edge_mlp_kernel(const _Float16* __restrict__ h16,
                                const float* __restrict__ efsrc, int efdim,
                                int efstride, const int* __restrict__ src,
                                const int* __restrict__ dst,
                                const _Float16* __restrict__ WU,
                                const _Float16* __restrict__ WV,
                                const _Float16* __restrict__ WE,
                                const float* __restrict__ b1,
                                const _Float16* __restrict__ W2,
                                const float* __restrict__ b2,
                                float* __restrict__ out_ef, int colbase) {
  __shared__ __attribute__((aligned(16))) _Float16 hid[4][16][kEHid];
  int wid  = threadIdx.x >> 5;
  int lane = threadIdx.x & 31;
  int M = lane & 15, hs = lane >> 4, N = lane & 15;
  int etile = blockIdx.x * 4 + wid;
  int edge = etile * 16 + M;
  int s = src[edge], d = dst[edge];
  const _Float16* urow = h16 + (size_t)s * kF;
  const _Float16* vrow = h16 + (size_t)d * kF;

  v16h au[4], av[4];
#pragma unroll
  for (int kt = 0; kt < 4; ++kt) {
    au[kt] = load_a16(urow, kt, hs);
    av[kt] = load_a16(vrow, kt, hs);
  }
  // ef contribution as a K-padded A fragment (only K < efdim non-zero).
  v16h aef;
#pragma unroll
  for (int i = 0; i < 16; ++i) aef[i] = (_Float16)0.0f;
  if (hs == 0) {
    const float* erow = efsrc + (size_t)edge * efstride;
    for (int i = 0; i < efdim; ++i) aef[i] = (_Float16)erow[i];
  }

#pragma unroll
  for (int nt = 0; nt < 2; ++nt) {
    float bval = b1[nt * 16 + N];
    v8f acc;
#pragma unroll
    for (int j = 0; j < 8; ++j) acc[j] = bval;
#pragma unroll
    for (int kt = 0; kt < 4; ++kt) {
      v16h bu = *reinterpret_cast<const v16h*>(WU + (size_t)(nt * 4 + kt) * 512 + lane * 16);
      v16h bw = *reinterpret_cast<const v16h*>(WV + (size_t)(nt * 4 + kt) * 512 + lane * 16);
      acc = wmma_f16(au[kt], bu, acc);
      acc = wmma_f16(av[kt], bw, acc);
    }
    v16h be = *reinterpret_cast<const v16h*>(WE + (size_t)nt * 512 + lane * 16);
    acc = wmma_f16(aef, be, acc);
#pragma unroll
    for (int j = 0; j < 8; ++j)
      hid[wid][j + 8 * hs][nt * 16 + N] = (_Float16)fmaxf(acc[j], 0.0f);
  }
  __syncthreads();

  v16h ah = load_a16(&hid[wid][M][0], 0, hs);   // hidden row as A fragment (K=32)
  v16h bw2 = *reinterpret_cast<const v16h*>(W2 + lane * 16);
  v8f acc2;
#pragma unroll
  for (int j = 0; j < 8; ++j) acc2[j] = (N < 4) ? b2[N] : 0.0f;
  acc2 = wmma_f16(ah, bw2, acc2);
  if (N < 4) {
#pragma unroll
    for (int j = 0; j < 8; ++j)
      out_ef[(size_t)(etile * 16 + j + 8 * hs) * 12 + colbase + N] = acc2[j];
  }
}

// ---------------------------------------------------------------------------
extern "C" void kernel_launch(void* const* d_in, const int* in_sizes, int n_in,
                              void* d_out, int out_size, void* d_ws, size_t ws_size,
                              hipStream_t stream) {
  (void)in_sizes; (void)n_in; (void)out_size; (void)ws_size;
  const float* inputs    = (const float*)d_in[0];
  const float* edge_feat = (const float*)d_in[1];
  const float* w_self1   = (const float*)d_in[2];
  const float* w_neigh1  = (const float*)d_in[3];
  const float* b_conv1   = (const float*)d_in[4];
  const float* eu1_wu    = (const float*)d_in[5];
  const float* eu1_wv    = (const float*)d_in[6];
  const float* eu1_we    = (const float*)d_in[7];
  const float* eu1_b     = (const float*)d_in[8];
  const float* eu1_w2    = (const float*)d_in[9];
  const float* eu1_b2    = (const float*)d_in[10];
  const float* w_self2   = (const float*)d_in[11];
  const float* w_neigh2  = (const float*)d_in[12];
  const float* b_conv2   = (const float*)d_in[13];
  const float* eu2_wu    = (const float*)d_in[14];
  const float* eu2_wv    = (const float*)d_in[15];
  const float* eu2_we    = (const float*)d_in[16];
  const float* eu2_b     = (const float*)d_in[17];
  const float* eu2_w2    = (const float*)d_in[18];
  const float* eu2_b2    = (const float*)d_in[19];
  const float* w_self3   = (const float*)d_in[20];
  const float* w_neigh3  = (const float*)d_in[21];
  const float* b_conv3   = (const float*)d_in[22];
  const int*   e_src     = (const int*)d_in[23];
  const int*   e_dst     = (const int*)d_in[24];

  float* out_h  = (float*)d_out;
  float* out_ef = out_h + (size_t)kNodes * kF;   // [E,12]

  uintptr_t ws = (uintptr_t)d_ws;
  auto take = [&](size_t bytes) -> void* {
    uintptr_t p = ws;
    ws += (bytes + 255) & ~(size_t)255;
    return (void*)p;
  };
  _Float16* h16a = (_Float16*)take((size_t)kNodes * kF * 2);
  _Float16* h16b = (_Float16*)take((size_t)kNodes * kF * 2);
  float*    agg  = (float*)take((size_t)kNodes * kF * 4);
  float*    rdeg = (float*)take((size_t)kNodes * 4);
  _Float16* pws1 = (_Float16*)take(32768);
  _Float16* pwn1 = (_Float16*)take(32768);
  _Float16* pws2 = (_Float16*)take(32768);
  _Float16* pwn2 = (_Float16*)take(32768);
  _Float16* pws3 = (_Float16*)take(32768);
  _Float16* pwn3 = (_Float16*)take(32768);
  _Float16* pwu1 = (_Float16*)take(8192);
  _Float16* pwv1 = (_Float16*)take(8192);
  _Float16* pwu2 = (_Float16*)take(8192);
  _Float16* pwv2 = (_Float16*)take(8192);
  _Float16* pwe1 = (_Float16*)take(2048);
  _Float16* pwe2 = (_Float16*)take(2048);
  _Float16* pw21 = (_Float16*)take(1024);
  _Float16* pw22 = (_Float16*)take(1024);

  auto pack = [&](const float* W, _Float16* dst, int K, int N, int Kp, int Np) {
    int total = Kp * Np;
    pack_b_f16<<<(total + 255) / 256, 256, 0, stream>>>(W, dst, K, N, Kp, Np);
  };
  pack(w_self1, pws1, 128, 128, 128, 128);
  pack(w_neigh1, pwn1, 128, 128, 128, 128);
  pack(w_self2, pws2, 128, 128, 128, 128);
  pack(w_neigh2, pwn2, 128, 128, 128, 128);
  pack(w_self3, pws3, 128, 128, 128, 128);
  pack(w_neigh3, pwn3, 128, 128, 128, 128);
  pack(eu1_wu, pwu1, 128, 32, 128, 32);
  pack(eu1_wv, pwv1, 128, 32, 128, 32);
  pack(eu2_wu, pwu2, 128, 32, 128, 32);
  pack(eu2_wv, pwv2, 128, 32, 128, 32);
  pack(eu1_we, pwe1, 4, 32, 32, 32);
  pack(eu2_we, pwe2, 8, 32, 32, 32);
  pack(eu1_w2, pw21, 32, 4, 32, 16);
  pack(eu2_w2, pw22, 32, 4, 32, 16);

  // Node features -> f16.
  {
    int n4 = kNodes * kF / 4;
    cvt_f32_to_f16<<<(n4 + 255) / 256, 256, 0, stream>>>(inputs, h16a, n4);
  }
  // Degrees (once; graph is fixed across layers).
  hipMemsetAsync(rdeg, 0, (size_t)kNodes * 4, stream);
  degree_kernel<<<(kEdges + 255) / 256, 256, 0, stream>>>(e_dst, rdeg, kEdges);
  rdeg_kernel<<<(kNodes + 255) / 256, 256, 0, stream>>>(rdeg, kNodes);
  // ef[:,0:4] = edge_feat.
  seed_edge_out<<<(kEdges * 4 + 255) / 256, 256, 0, stream>>>(edge_feat, out_ef,
                                                              kEdges * 4);

  const int sageBlocks = (kNodes / 16) * (kF / 16) / 4;  // 6250
  const int edgeBlocks = (kEdges / 16) / 4;              // 12500

  // ---- Layer 1 ----
  hipMemsetAsync(agg, 0, (size_t)kNodes * kF * 4, stream);
  scatter_mean_kernel<<<kEdges, 128, 0, stream>>>(h16a, e_src, e_dst, agg);
  sage_wmma_kernel<true, true><<<sageBlocks, 128, 0, stream>>>(
      h16a, agg, rdeg, pws1, pwn1, b_conv1, h16b, nullptr);
  // ---- Edge update 1: ef[:,4:8] ----
  edge_mlp_kernel<<<edgeBlocks, 128, 0, stream>>>(
      h16b, edge_feat, 4, 4, e_src, e_dst, pwu1, pwv1, pwe1, eu1_b, pw21, eu1_b2,
      out_ef, 4);
  // ---- Layer 2 ----
  hipMemsetAsync(agg, 0, (size_t)kNodes * kF * 4, stream);
  scatter_mean_kernel<<<kEdges, 128, 0, stream>>>(h16b, e_src, e_dst, agg);
  sage_wmma_kernel<true, true><<<sageBlocks, 128, 0, stream>>>(
      h16b, agg, rdeg, pws2, pwn2, b_conv2, h16a, nullptr);
  // ---- Edge update 2: reads ef[:,0:8], writes ef[:,8:12] ----
  edge_mlp_kernel<<<edgeBlocks, 128, 0, stream>>>(
      h16a, out_ef, 8, 12, e_src, e_dst, pwu2, pwv2, pwe2, eu2_b, pw22, eu2_b2,
      out_ef, 8);
  // ---- Layer 3 (no relu, f32 out) ----
  hipMemsetAsync(agg, 0, (size_t)kNodes * kF * 4, stream);
  scatter_mean_kernel<<<kEdges, 128, 0, stream>>>(h16a, e_src, e_dst, agg);
  sage_wmma_kernel<false, false><<<sageBlocks, 128, 0, stream>>>(
      h16a, agg, rdeg, pws3, pwn3, b_conv3, nullptr, out_h);
}
